// Block_47665547051044
// MI455X (gfx1250) — matrix operand order repaired and checked by reading, not compiled, not run on hardware
//
#include <hip/hip_runtime.h>

// ---------------------------------------------------------------------------
// Transformer block for MI455X (gfx1250): bf16 WMMA GEMMs + flash attention.
// ---------------------------------------------------------------------------

typedef __attribute__((ext_vector_type(16))) __bf16 v16bf;
typedef __attribute__((ext_vector_type(8)))  float  v8f;

#define BDIM 256
#define B_   2
#define T_   2048
#define C_   1024
#define H_   16
#define HS_  64
#define FF_  4096
#define BT_  (B_ * T_)

union FragBF { v16bf v; uint4 u[2]; __bf16 h[16]; };
union Pack8  { uint4 u; __bf16 h[8]; unsigned short s[8]; };

__device__ __forceinline__ v8f wmma_bf16(const FragBF& a, const FragBF& b, v8f c) {
  return __builtin_amdgcn_wmma_f32_16x16x32_bf16(
      /*neg_a=*/false, a.v, /*neg_b=*/false, b.v,
      /*c_mod=*/(short)0, c, /*reuse_a=*/false, /*reuse_b=*/false);
}

// Async global->LDS staging if the toolchain exposes it (gfx1250 ASYNCcnt path)
#if defined(__HIP_DEVICE_COMPILE__) && \
    __has_builtin(__builtin_amdgcn_global_load_async_to_lds_b128)
#define GEMM_ASYNC 1
#else
#define GEMM_ASYNC 0
#endif

#define AS1 __attribute__((address_space(1)))
#define AS3 __attribute__((address_space(3)))
typedef int v4i_gnu __attribute__((__vector_size__(4 * sizeof(int))));

__device__ __forceinline__ void wait_async0() {
#if defined(__HIP_DEVICE_COMPILE__)
#if __has_builtin(__builtin_amdgcn_s_wait_asynccnt)
  __builtin_amdgcn_s_wait_asynccnt(0);
#else
  asm volatile("s_wait_asynccnt 0x0" ::: "memory");
#endif
#endif
}

// ---------------------------------------------------------------------------
// Tiled bf16 GEMM:  C[M,N] = relu?(A[M,K] * B[K,N] + bias[N])
// Block tile 128x128, 8 waves, each wave computes 32x64 (8 WMMAs per K-step
// of 32). A staged to LDS (async path when available); B staged transposed
// via register pair-packing + ds_store_b32. M,N multiples of 128; K of 32.
// ---------------------------------------------------------------------------
__global__ __launch_bounds__(BDIM)
void gemm_bf16_kernel(const __bf16* __restrict__ A, const __bf16* __restrict__ B,
                      const float* __restrict__ bias, void* __restrict__ Cout,
                      int M, int N, int K, int do_relu, int out_bf16)
{
  __shared__ __align__(16) __bf16 As[128][40];  // [m][k], pad 8 halves
  __shared__ __align__(16) __bf16 Bs[128][40];  // [n][k] (transposed), pad 8

  const int tid  = threadIdx.x;
  const int w    = tid >> 5;
  const int lane = tid & 31;
  const int g    = lane >> 4;        // lane group (0 or 1)
  const int col  = lane & 15;
  const int m0   = blockIdx.y * 128;
  const int n0   = blockIdx.x * 128;
  const int wr   = (w & 3) * 32;     // wave row offset in block tile
  const int wc   = (w >> 2) * 64;    // wave col offset in block tile

  // Cooperative staging assignments
  const int arow = tid >> 1, ac = (tid & 1) * 16;        // A: 1 row x 16 halves
  const int bkp  = (tid >> 4) * 2,  bnc = (tid & 15) * 8; // B: 2 k-rows x 8 n

  v8f acc[2][4];
#pragma unroll
  for (int i = 0; i < 2; ++i)
#pragma unroll
    for (int j = 0; j < 4; ++j)
      acc[i][j] = (v8f){0.f,0.f,0.f,0.f,0.f,0.f,0.f,0.f};

  for (int k0 = 0; k0 < K; k0 += 32) {
    // ---- stage A tile: 128 rows x 32 halves ----
    const __bf16* agp = A + (size_t)(m0 + arow) * K + k0 + ac;
#if GEMM_ASYNC
    __builtin_amdgcn_global_load_async_to_lds_b128(
        (AS1 v4i_gnu*)agp, (AS3 v4i_gnu*)&As[arow][ac], 0, 0);
    __builtin_amdgcn_global_load_async_to_lds_b128(
        (AS1 v4i_gnu*)(agp + 8), (AS3 v4i_gnu*)&As[arow][ac + 8], 0, 0);
#else
    *(uint4*)&As[arow][ac]     = *(const uint4*)agp;
    *(uint4*)&As[arow][ac + 8] = *(const uint4*)(agp + 8);
#endif
    // ---- stage B tile transposed: pack (k,k+1) pairs, ds_store_b32 ----
    Pack8 r0, r1;
    r0.u = *(const uint4*)(B + (size_t)(k0 + bkp) * N + n0 + bnc);
    r1.u = *(const uint4*)(B + (size_t)(k0 + bkp + 1) * N + n0 + bnc);
#pragma unroll
    for (int i = 0; i < 8; ++i) {
      const unsigned int pair = (unsigned int)r0.s[i] |
                                ((unsigned int)r1.s[i] << 16);
      *(unsigned int*)&Bs[bnc + i][bkp] = pair;
    }
#if GEMM_ASYNC
    wait_async0();
#endif
    __syncthreads();

    // ---- fragments + 8 WMMAs ----
    FragBF fa[2], fb[4];
#pragma unroll
    for (int i = 0; i < 2; ++i) {
      fa[i].u[0] = *(const uint4*)&As[wr + i * 16 + col][g * 8];
      fa[i].u[1] = *(const uint4*)&As[wr + i * 16 + col][16 + g * 8];
    }
#pragma unroll
    for (int j = 0; j < 4; ++j) {
      fb[j].u[0] = *(const uint4*)&Bs[wc + j * 16 + col][g * 8];
      fb[j].u[1] = *(const uint4*)&Bs[wc + j * 16 + col][16 + g * 8];
    }
#pragma unroll
    for (int i = 0; i < 2; ++i)
#pragma unroll
      for (int j = 0; j < 4; ++j)
        acc[i][j] = wmma_bf16(fa[i], fb[j], acc[i][j]);
    __syncthreads();
  }

  // ---- epilogue: D layout lane->col, VGPR r -> row g*8+r ----
#pragma unroll
  for (int i = 0; i < 2; ++i) {
#pragma unroll
    for (int j = 0; j < 4; ++j) {
      const int n = n0 + wc + j * 16 + col;
      const float bv = bias ? bias[n] : 0.f;
#pragma unroll
      for (int r = 0; r < 8; ++r) {
        const int m = m0 + wr + i * 16 + g * 8 + r;
        float val = acc[i][j][r] + bv;
        if (do_relu) val = fmaxf(val, 0.f);
        if (out_bf16) ((__bf16*)Cout)[(size_t)m * N + n] = (__bf16)val;
        else          ((float*)Cout)[(size_t)m * N + n] = val;
      }
    }
  }
}

// ---------------------------------------------------------------------------
// Flash attention (causal). q/k/v/o: bf16 [B,T,H,HS] (head-interleaved [B,T,C]).
// Block = 8 waves = 128 queries of one (b,h). Key tiles of 32 staged in LDS.
// Computes S^T = K*Q^T (softmax stats per-lane after one xor-16 merge) and
// out^T = V^T * P^T (rescale + 1/l normalization are per-lane scalars).
// ---------------------------------------------------------------------------
__global__ __launch_bounds__(BDIM)
void attn_kernel(const __bf16* __restrict__ q, const __bf16* __restrict__ k,
                 const __bf16* __restrict__ v, __bf16* __restrict__ o)
{
  __shared__ __align__(16) __bf16 Ks[32][72];  // [key][hs], pad 8
  __shared__ __align__(16) __bf16 Vt[64][40];  // [hs][key] transposed, pad 8

  const int tid  = threadIdx.x;
  const int w    = tid >> 5;
  const int lane = tid & 31;
  const int g    = lane >> 4;
  const int col  = lane & 15;
  const int b    = blockIdx.z;
  const int hh   = blockIdx.y;
  const int q0w  = blockIdx.x * 128 + w * 16;
  const int query = q0w + col;               // this lane's query (per group)

  // Q^T B-fragments for the two HS k-steps: halves i<8 -> hs = s*32 + g*8 + i, i>=8 -> s*32 + 16 + g*8 + (i-8)
  const __bf16* qrow = q + ((size_t)(b * T_ + query)) * C_ + hh * HS_;
  FragBF qf[2];
#pragma unroll
  for (int s = 0; s < 2; ++s) {
    qf[s].u[0] = *(const uint4*)(qrow + s * 32 + g * 8);
    qf[s].u[1] = *(const uint4*)(qrow + s * 32 + 16 + g * 8);
  }

  const float NEG_INF = -__builtin_inff();
  float m_run = NEG_INF, l_run = 0.f;
  v8f ot[4];
#pragma unroll
  for (int t = 0; t < 4; ++t) ot[t] = (v8f){0.f,0.f,0.f,0.f,0.f,0.f,0.f,0.f};

  const int krow = tid >> 3, kc = (tid & 7) * 8;
  const int ntiles = blockIdx.x * 4 + 4;     // covers keys <= block max query

  for (int kt = 0; kt < ntiles; ++kt) {
    const int kb = kt * 32;
    // Stage K tile (32 keys x 64 hs) and V tile transposed
    const size_t rbase = ((size_t)(b * T_ + kb + krow)) * C_ + hh * HS_ + kc;
    *(uint4*)&Ks[krow][kc] = *(const uint4*)(k + rbase);
    Pack8 pv; pv.u = *(const uint4*)(v + rbase);
#pragma unroll
    for (int i = 0; i < 8; ++i) Vt[kc + i][krow] = pv.h[i];
    __syncthreads();

    if (kb <= q0w + 15) {                    // wave-uniform activity test
      float s0[8], s1[8];
#pragma unroll
      for (int tt = 0; tt < 2; ++tt) {
        v8f sa = (v8f){0.f,0.f,0.f,0.f,0.f,0.f,0.f,0.f};
#pragma unroll
        for (int s = 0; s < 2; ++s) {
          FragBF ka;                         // K-tile A frag (16 keys x 32 hs)
          ka.u[0] = *(const uint4*)&Ks[tt * 16 + col][s * 32 + g * 8];
          ka.u[1] = *(const uint4*)&Ks[tt * 16 + col][s * 32 + 16 + g * 8];
          sa = wmma_bf16(ka, qf[s], sa);
        }
        float* dst = tt ? s1 : s0;
#pragma unroll
        for (int r = 0; r < 8; ++r) {
          const int key = kb + tt * 16 + g * 8 + r;
          dst[r] = (key <= query) ? sa[r] * 0.125f : NEG_INF;  // HS^-0.5
        }
      }
      // Online softmax (per query == per lane after xor-16 merge)
      float mt = NEG_INF;
#pragma unroll
      for (int r = 0; r < 8; ++r) mt = fmaxf(mt, fmaxf(s0[r], s1[r]));
      mt = fmaxf(mt, __shfl_xor(mt, 16, 32));
      const float mnew = fmaxf(m_run, mt);
      const bool dead = (mnew == NEG_INF);
      const float alpha = dead ? 1.f : __expf(m_run - mnew);
      float p0[8], p1[8], ls = 0.f;
#pragma unroll
      for (int r = 0; r < 8; ++r) {
        p0[r] = dead ? 0.f : __expf(s0[r] - mnew);
        p1[r] = dead ? 0.f : __expf(s1[r] - mnew);
        ls += p0[r] + p1[r];
      }
      ls += __shfl_xor(ls, 16, 32);
      l_run = l_run * alpha + ls;
      m_run = mnew;
#pragma unroll
      for (int t = 0; t < 4; ++t)
#pragma unroll
        for (int r = 0; r < 8; ++r) ot[t][r] *= alpha;

      // Build P^T B-frag (K = 32 keys): cross-group xor-16 exchange
      FragBF pf;
#pragma unroll
      for (int r = 0; r < 8; ++r) {
        const float o0 = __shfl_xor(p0[r], 16, 32);
        const float o1 = __shfl_xor(p1[r], 16, 32);
        const float lo = g ? o1 : p0[r];     // keys g*16 + r
        const float hi = g ? p1[r] : o0;     // keys g*16 + 8 + r
        pf.h[r]     = (__bf16)lo;
        pf.h[8 + r] = (__bf16)hi;
      }
      // out^T += V^T * P^T  (four 16-row HS sub-tiles)
#pragma unroll
      for (int t = 0; t < 4; ++t) {
        FragBF va;
        va.u[0] = *(const uint4*)&Vt[t * 16 + col][g * 16];
        va.u[1] = *(const uint4*)&Vt[t * 16 + col][g * 16 + 8];
        ot[t] = wmma_bf16(va, pf, ot[t]);
      }
    }
    __syncthreads();
  }

  // Epilogue: normalize per query (per lane) and store contiguous 8 halves
  const float inv = 1.f / l_run;
  __bf16* op = o + ((size_t)(b * T_ + query)) * C_ + hh * HS_;
#pragma unroll
  for (int t = 0; t < 4; ++t) {
    Pack8 pk;
#pragma unroll
    for (int r = 0; r < 8; ++r) pk.h[r] = (__bf16)(ot[t][r] * inv);
    *(uint4*)(op + t * 16 + g * 8) = pk.u;
  }
}

// ---------------------------------------------------------------------------
// LayerNorm over C=1024 per row; block per row, 4 elements per thread.
// ---------------------------------------------------------------------------
__global__ __launch_bounds__(BDIM)
void ln_kernel(const float* __restrict__ in, const float* __restrict__ gw,
               const float* __restrict__ bw, void* __restrict__ out, int out_bf16)
{
  __shared__ float red[BDIM];
  const int row = blockIdx.x, tid = threadIdx.x;
  const float* x = in + (size_t)row * C_;
  float vals[4], s = 0.f;
#pragma unroll
  for (int i = 0; i < 4; ++i) { vals[i] = x[tid + i * BDIM]; s += vals[i]; }
  red[tid] = s; __syncthreads();
  for (int st = BDIM / 2; st > 0; st >>= 1) {
    if (tid < st) red[tid] += red[tid + st];
    __syncthreads();
  }
  const float mean = red[0] / (float)C_;
  __syncthreads();
  float s2 = 0.f;
#pragma unroll
  for (int i = 0; i < 4; ++i) { const float d = vals[i] - mean; s2 += d * d; }
  red[tid] = s2; __syncthreads();
  for (int st = BDIM / 2; st > 0; st >>= 1) {
    if (tid < st) red[tid] += red[tid + st];
    __syncthreads();
  }
  const float rstd = rsqrtf(red[0] / (float)C_ + 1e-5f);
#pragma unroll
  for (int i = 0; i < 4; ++i) {
    const int c = tid + i * BDIM;
    const float y = (vals[i] - mean) * rstd * gw[c] + bw[c];
    if (out_bf16) ((__bf16*)out)[(size_t)row * C_ + c] = (__bf16)y;
    else          ((float*)out)[(size_t)row * C_ + c] = y;
  }
}

// ---------------------------------------------------------------------------
// Packing kernels
// ---------------------------------------------------------------------------
__global__ void pack_bf16_kernel(const float* __restrict__ in,
                                 __bf16* __restrict__ out, int n)
{
  const int i = blockIdx.x * blockDim.x + threadIdx.x;
  if (i < n) out[i] = (__bf16)in[i];
}

// [H,C,HS] fp32 -> combined [K=C, N=C] bf16 with N = h*HS + d
__global__ void pack_headw_kernel(const float* __restrict__ in,
                                  __bf16* __restrict__ out)
{
  const int i = blockIdx.x * blockDim.x + threadIdx.x;
  if (i >= C_ * C_) return;
  const int kk = i / C_, n = i % C_;
  const int hh = n >> 6, d = n & 63;
  out[i] = (__bf16)in[(size_t)hh * C_ * HS_ + (size_t)kk * HS_ + d];
}

// ---------------------------------------------------------------------------
extern "C" void kernel_launch(void* const* d_in, const int* in_sizes, int n_in,
                              void* d_out, int out_size, void* d_ws, size_t ws_size,
                              hipStream_t stream)
{
  (void)in_sizes; (void)n_in; (void)out_size; (void)ws_size;
  const float* x   = (const float*)d_in[0];
  const float* Wq  = (const float*)d_in[1];
  const float* bq  = (const float*)d_in[2];   // [H,HS] flat == [C]
  const float* Wk  = (const float*)d_in[3];
  const float* bk  = (const float*)d_in[4];
  const float* Wv  = (const float*)d_in[5];
  const float* bv  = (const float*)d_in[6];
  const float* Wp  = (const float*)d_in[7];
  const float* bp  = (const float*)d_in[8];
  const float* W1  = (const float*)d_in[9];
  const float* b1  = (const float*)d_in[10];
  const float* W2  = (const float*)d_in[11];
  const float* b2  = (const float*)d_in[12];
  const float* g1  = (const float*)d_in[13];
  const float* be1 = (const float*)d_in[14];
  const float* g2  = (const float*)d_in[15];
  const float* be2 = (const float*)d_in[16];

  char* wp = (char*)d_ws;
  auto alloc = [&](size_t bytes) -> void* {
    void* r = (void*)wp; wp += (bytes + 255) & ~(size_t)255; return r;
  };
  __bf16* xb   = (__bf16*)alloc((size_t)BT_ * C_ * 2);
  __bf16* wqb  = (__bf16*)alloc((size_t)C_ * C_ * 2);
  __bf16* wkb  = (__bf16*)alloc((size_t)C_ * C_ * 2);
  __bf16* wvb  = (__bf16*)alloc((size_t)C_ * C_ * 2);
  __bf16* wpb  = (__bf16*)alloc((size_t)C_ * C_ * 2);
  __bf16* w1b  = (__bf16*)alloc((size_t)C_ * FF_ * 2);
  __bf16* w2b  = (__bf16*)alloc((size_t)FF_ * C_ * 2);
  __bf16* qb   = (__bf16*)alloc((size_t)BT_ * C_ * 2);
  __bf16* kb   = (__bf16*)alloc((size_t)BT_ * C_ * 2);
  __bf16* vb   = (__bf16*)alloc((size_t)BT_ * C_ * 2);
  __bf16* ao   = (__bf16*)alloc((size_t)BT_ * C_ * 2);
  float*  hbuf = (float*) alloc((size_t)BT_ * C_ * 4);
  __bf16* hn   = (__bf16*)alloc((size_t)BT_ * C_ * 2);
  __bf16* fb   = (__bf16*)alloc((size_t)BT_ * FF_ * 2);
  float*  f2   = (float*) alloc((size_t)BT_ * C_ * 4);

  // ---- pack inputs to bf16 ----
  pack_bf16_kernel<<<(BT_ * C_ + 255) / 256, 256, 0, stream>>>(x, xb, BT_ * C_);
  pack_headw_kernel<<<(C_ * C_ + 255) / 256, 256, 0, stream>>>(Wq, wqb);
  pack_headw_kernel<<<(C_ * C_ + 255) / 256, 256, 0, stream>>>(Wk, wkb);
  pack_headw_kernel<<<(C_ * C_ + 255) / 256, 256, 0, stream>>>(Wv, wvb);
  pack_bf16_kernel<<<(C_ * C_ + 255) / 256, 256, 0, stream>>>(Wp, wpb, C_ * C_);
  pack_bf16_kernel<<<(C_ * FF_ + 255) / 256, 256, 0, stream>>>(W1, w1b, C_ * FF_);
  pack_bf16_kernel<<<(FF_ * C_ + 255) / 256, 256, 0, stream>>>(W2, w2b, FF_ * C_);

  // ---- QKV projections (outputs head-interleaved [B,T,H,HS] bf16) ----
  dim3 gqkv(C_ / 128, BT_ / 128);
  gemm_bf16_kernel<<<gqkv, BDIM, 0, stream>>>(xb, wqb, bq, qb, BT_, C_, C_, 0, 1);
  gemm_bf16_kernel<<<gqkv, BDIM, 0, stream>>>(xb, wkb, bk, kb, BT_, C_, C_, 0, 1);
  gemm_bf16_kernel<<<gqkv, BDIM, 0, stream>>>(xb, wvb, bv, vb, BT_, C_, C_, 0, 1);

  // ---- causal flash attention ----
  dim3 ga(T_ / 128, H_, B_);
  attn_kernel<<<ga, BDIM, 0, stream>>>(qb, kb, vb, ao);

  // ---- output projection + LN1 ----
  dim3 gp(C_ / 128, BT_ / 128);
  gemm_bf16_kernel<<<gp, BDIM, 0, stream>>>(ao, wpb, bp, hbuf, BT_, C_, C_, 0, 0);
  ln_kernel<<<BT_, BDIM, 0, stream>>>(hbuf, g1, be1, hn, 1);

  // ---- feed-forward: relu(h@W1+b1) -> relu(f@W2+b2) -> LN2 ----
  dim3 gf1(FF_ / 128, BT_ / 128);
  gemm_bf16_kernel<<<gf1, BDIM, 0, stream>>>(hn, w1b, b1, fb, BT_, FF_, C_, 1, 1);
  dim3 gf2(C_ / 128, BT_ / 128);
  gemm_bf16_kernel<<<gf2, BDIM, 0, stream>>>(fb, w2b, b2, f2, BT_, C_, FF_, 1, 0);
  ln_kernel<<<BT_, BDIM, 0, stream>>>(f2, g2, be2, d_out, 0);
}
